// CifPredictorV2_13915694039446
// MI455X (gfx1250) — matrix operand order repaired and checked by reading, compiled, not run on hardware
//
#include <hip/hip_runtime.h>
#include <hip/hip_bf16.h>

// ---------------------------------------------------------------------------
// CifPredictorV2 for MI455X (gfx1250, wave32, WMMA + async Global->LDS)
// Pipeline:
//   K0 zero     : zero acoustic_embeds + token_num in d_out, zero alpha_acc
//   K1 xpad     : hidden f32 -> bf16, zero-padded [B, T+2, D] (im2col-free A)
//   K2 wprep    : conv_w [D,D,K] f32 -> Wb bf16 column-major [N=512][K=1536]
//   K3 gemm     : A[32000x1536] x Wb -> relu(+bias), fused out_w dot.
//                 Weights double-buffered into LDS via async global->LDS
//                 (ASYNCcnt), compute via v_wmma_f32_16x16x32_bf16.
//   K4 alpha    : sigmoid(acc+out_b)*mask -> alphas, atomic token_num
//   K5 cif      : sequential CIF scan, compacting fired frames
// ---------------------------------------------------------------------------

typedef __bf16 bf16x16 __attribute__((ext_vector_type(16)));
typedef __bf16 bf16x8  __attribute__((ext_vector_type(8)));
typedef float  f32x8   __attribute__((ext_vector_type(8)));
typedef int    v4i     __attribute__((ext_vector_type(4)));

#define BB 16
#define TT 2000
#define DD 512
#define TP 2002           // T + 2 (zero pad rows)
#define KDIM 1536         // 3 * 512
#define MROWS 32000       // B * T

#define KC 128            // K-chunk staged per LDS buffer
#define KCP 136           // padded per-column stride (272 B == 16 mod 256 ->
                          // the 16 lanes of a ds_load_b128 hit disjoint banks)
#define NCHUNK (KDIM / KC)  // 12

union Frag { bf16x16 v; bf16x8 h[2]; };

static __device__ __forceinline__ bf16x8 ld8(const unsigned short* p) {
  return *(const bf16x8*)(const void*)p;
}

static __device__ __forceinline__ unsigned short f32_to_bf16(float f) {
  unsigned int u = __float_as_uint(f);
  u = u + 0x7FFFu + ((u >> 16) & 1u);   // round-to-nearest-even
  return (unsigned short)(u >> 16);
}

// ---- async Global->LDS (gfx1250). Guarded so either toolchain compiles. ----
#if __has_builtin(__builtin_amdgcn_global_load_async_to_lds_b128)
#define HAVE_ASYNC_LDS 1
#else
#define HAVE_ASYNC_LDS 0
#endif

static __device__ __forceinline__ void async_ld16(const unsigned short* g,
                                                  unsigned short* l) {
#if HAVE_ASYNC_LDS
  __builtin_amdgcn_global_load_async_to_lds_b128(
      (__attribute__((address_space(1))) v4i*)(void*)g,
      (__attribute__((address_space(3))) v4i*)(void*)l,
      0, 0);
#else
  *(bf16x8*)(void*)l = *(const bf16x8*)(const void*)g;  // regs + ds_store
#endif
}

static __device__ __forceinline__ void wait_async0() {
#if HAVE_ASYNC_LDS
#if __has_builtin(__builtin_amdgcn_s_wait_asynccnt)
  __builtin_amdgcn_s_wait_asynccnt(0);
#else
  asm volatile("s_wait_asynccnt 0" ::: "memory");
#endif
#endif
}

// --------------------------------- K0 --------------------------------------
__global__ void k_zero(float* __restrict__ p, int n) {
  int i = blockIdx.x * blockDim.x + threadIdx.x;
  int stride = gridDim.x * blockDim.x;
  for (; i < n; i += stride) p[i] = 0.0f;
}

// --------------------------------- K1 --------------------------------------
__global__ void k_xpad(const float* __restrict__ hidden,
                       unsigned short* __restrict__ Xpad) {
  const int n = BB * TP * DD;
  int i = blockIdx.x * blockDim.x + threadIdx.x;
  int stride = gridDim.x * blockDim.x;
  for (; i < n; i += stride) {
    int b   = i / (TP * DD);
    int rem = i - b * (TP * DD);
    int tp  = rem >> 9;           // / 512
    int d   = rem & 511;
    float v = 0.0f;
    if (tp >= 1 && tp <= TT)
      v = hidden[((size_t)b * TT + (tp - 1)) * DD + d];
    Xpad[i] = f32_to_bf16(v);
  }
}

// --------------------------------- K2 --------------------------------------
// Wb[o][k*512 + i] = conv_w[o][i][k]   (column-major B: each lane's WMMA
// B-fragment is 32 contiguous bytes)
__global__ void k_wprep(const float* __restrict__ conv_w,
                        unsigned short* __restrict__ Wb) {
  const int n = DD * KDIM;
  int i = blockIdx.x * blockDim.x + threadIdx.x;
  int stride = gridDim.x * blockDim.x;
  for (; i < n; i += stride) {
    int o  = i / KDIM;
    int kk = i - o * KDIM;
    int k  = kk >> 9;             // / 512
    int ii = kk & 511;
    Wb[i] = f32_to_bf16(conv_w[((size_t)o * DD + ii) * 3 + k]);
  }
}

// --------------------------------- K3 --------------------------------------
// Grid: (250, 8). 256 threads = 8 waves. Each wave: 16 rows x 64 cols.
// All 8 waves share the same 64 weight columns -> stage them in LDS once per
// K-chunk (async, double-buffered) instead of 8x redundant VMEM loads.
__global__ __launch_bounds__(256)
void k_gemm(const unsigned short* __restrict__ Xpad,
            const unsigned short* __restrict__ Wb,
            const float* __restrict__ conv_b,
            const float* __restrict__ out_w,
            float* __restrict__ alpha_acc) {
  extern __shared__ unsigned short sB[];   // 2 x 64 x KCP bf16

  const int tid  = threadIdx.x;
  const int wave = tid >> 5;
  const int lane = tid & 31;
  const int lh   = lane >> 4;     // 0 or 1 (K-half select)
  const int ll   = lane & 15;

  const int rowbase = blockIdx.x * 128 + wave * 16;
  const int row = rowbase + ll;
  const int b = row / TT;
  const int t = row - b * TT;
  // A row (b,t) = 1536 contiguous bf16 at Xpad[b, t, 0]
  const unsigned short* aptr = Xpad + ((size_t)(b * TP + t)) * DD + lh * 16;

  const int nbase = blockIdx.y * 64;

  f32x8 acc0 = {0,0,0,0,0,0,0,0};
  f32x8 acc1 = {0,0,0,0,0,0,0,0};
  f32x8 acc2 = {0,0,0,0,0,0,0,0};
  f32x8 acc3 = {0,0,0,0,0,0,0,0};

  // cooperative async stage of one K-chunk of B into LDS buffer `buf`
  auto issue_chunk = [&](int buf, int kc) {
    unsigned short* dst = sB + buf * (64 * KCP);
    // 64 cols x 128 K x 2B = 16KB = 1024 x 16B; 256 threads -> 4 each
#pragma unroll
    for (int it = 0; it < 4; ++it) {
      int li = it * 256 + tid;
      int c  = li >> 4;             // column 0..63
      int k8 = (li & 15) << 3;      // K offset in elements, 16B granules
      async_ld16(Wb + (size_t)(nbase + c) * KDIM + kc + k8,
                 dst + c * KCP + k8);
    }
  };

  issue_chunk(0, 0);
  wait_async0();
  __syncthreads();

  for (int ch = 0; ch < NCHUNK; ++ch) {
    const int cur = ch & 1;
    if (ch + 1 < NCHUNK) issue_chunk(cur ^ 1, (ch + 1) * KC);

    const unsigned short* sb = sB + cur * (64 * KCP);
    const unsigned short* b0 = sb + (ll     ) * KCP + lh * 16;
    const unsigned short* b1 = sb + (ll + 16) * KCP + lh * 16;
    const unsigned short* b2 = sb + (ll + 32) * KCP + lh * 16;
    const unsigned short* b3 = sb + (ll + 48) * KCP + lh * 16;
    const unsigned short* ap = aptr + ch * KC;

#pragma unroll
    for (int kk = 0; kk < KC; kk += 32) {
      __builtin_prefetch((const void*)(ap + kk + 256), 0, 0);
      Frag a;
      a.h[0] = ld8(ap + kk);
      a.h[1] = ld8(ap + kk + 8);
      Frag f0, f1, f2, f3;
      f0.h[0] = ld8(b0 + kk); f0.h[1] = ld8(b0 + kk + 8);
      f1.h[0] = ld8(b1 + kk); f1.h[1] = ld8(b1 + kk + 8);
      f2.h[0] = ld8(b2 + kk); f2.h[1] = ld8(b2 + kk + 8);
      f3.h[0] = ld8(b3 + kk); f3.h[1] = ld8(b3 + kk + 8);

      acc0 = __builtin_amdgcn_wmma_f32_16x16x32_bf16(false, a.v, false, f0.v,
                                                     (short)0, acc0, false, false);
      acc1 = __builtin_amdgcn_wmma_f32_16x16x32_bf16(false, a.v, false, f1.v,
                                                     (short)0, acc1, false, false);
      acc2 = __builtin_amdgcn_wmma_f32_16x16x32_bf16(false, a.v, false, f2.v,
                                                     (short)0, acc2, false, false);
      acc3 = __builtin_amdgcn_wmma_f32_16x16x32_bf16(false, a.v, false, f3.v,
                                                     (short)0, acc3, false, false);
    }

    if (ch + 1 < NCHUNK) wait_async0();
    __syncthreads();   // all waves done reading buf[cur]; next buffer ready
  }

  // Epilogue: yv = relu(acc + bias); partial = sum_o yv * out_w[o]
  // C layout: element e of lane L = y[M = e + 8*(L>>4), N = nblk*16 + (L&15)]
  float bias0 = conv_b[nbase + ll];
  float bias1 = conv_b[nbase + 16 + ll];
  float bias2 = conv_b[nbase + 32 + ll];
  float bias3 = conv_b[nbase + 48 + ll];
  float w0 = out_w[nbase + ll];
  float w1 = out_w[nbase + 16 + ll];
  float w2 = out_w[nbase + 32 + ll];
  float w3 = out_w[nbase + 48 + ll];

  float p[8];
#pragma unroll
  for (int e = 0; e < 8; ++e) {
    float y0 = acc0[e] + bias0; y0 = y0 > 0.0f ? y0 : 0.0f;
    float y1 = acc1[e] + bias1; y1 = y1 > 0.0f ? y1 : 0.0f;
    float y2 = acc2[e] + bias2; y2 = y2 > 0.0f ? y2 : 0.0f;
    float y3 = acc3[e] + bias3; y3 = y3 > 0.0f ? y3 : 0.0f;
    p[e] = y0 * w0 + y1 * w1 + y2 * w2 + y3 * w3;
  }
  // reduce across the 16 lanes within each half-wave (rows differ per half)
#pragma unroll
  for (int m = 1; m < 16; m <<= 1) {
#pragma unroll
    for (int e = 0; e < 8; ++e)
      p[e] += __shfl_xor(p[e], m, 32);
  }
  if (ll == 0) {
#pragma unroll
    for (int e = 0; e < 8; ++e)
      atomicAdd(&alpha_acc[rowbase + e + 8 * lh], p[e]);
  }
}

// --------------------------------- K4 --------------------------------------
__global__ void k_alpha(const float* __restrict__ acc,
                        const float* __restrict__ mask,
                        const float* __restrict__ out_b,
                        float* __restrict__ alphas,
                        float* __restrict__ token) {
  int r = blockIdx.x * blockDim.x + threadIdx.x;
  if (r >= MROWS) return;
  int b = r / TT;
  int t = r - b * TT;
  float x = acc[r] + out_b[0];
  float a = 1.0f / (1.0f + __expf(-x));   // relu(sigmoid*1 - 0) == sigmoid
  a *= mask[b * TT + t];
  alphas[r] = a;
  atomicAdd(&token[b], a);
}

// --------------------------------- K5 --------------------------------------
// One block per batch row, 512 threads = one channel each; sequential T scan.
__global__ __launch_bounds__(512)
void k_cif(const float* __restrict__ hidden,
           const float* __restrict__ alphas,
           float* __restrict__ acoustic,
           float* __restrict__ peak) {
  __shared__ float s_cur, s_rem;
  __shared__ int   s_fire, s_dest;
  const int b = blockIdx.x;
  const int d = threadIdx.x;
  float integ = 0.0f;
  int   cnt   = 0;
  float frame = 0.0f;
  const float* hb = hidden + (size_t)b * TT * DD;
  const float* ab = alphas + b * TT;
  float* ob = acoustic + (size_t)b * TT * DD;
  float* pb = peak + b * TT;

  if (d == 0) { s_cur = 0.0f; s_rem = 0.0f; s_fire = 0; s_dest = 0; }
  __syncthreads();

  for (int t = 0; t < TT; ++t) {
    float h = hb[t * DD + d];
    if (d == 0) {
      float a    = ab[t];
      float dist = 1.0f - integ;
      integ += a;
      int fire  = integ >= 1.0f;
      float cur = fire ? dist : a;
      s_cur  = cur;
      s_rem  = a - cur;
      s_fire = fire;
      pb[t]  = integ;              // cif_peak = post-add integrate
      if (fire) { integ -= 1.0f; s_dest = cnt; cnt++; }
    }
    __syncthreads();
    float cur  = s_cur;
    float rem  = s_rem;
    int   fire = s_fire;
    int   dest = s_dest;
    frame += cur * h;
    if (fire) {
      ob[(size_t)dest * DD + d] = frame;   // compacted store
      frame = rem * h;
    }
    __syncthreads();
  }
}

// ---------------------------------------------------------------------------
extern "C" void kernel_launch(void* const* d_in, const int* in_sizes, int n_in,
                              void* d_out, int out_size, void* d_ws, size_t ws_size,
                              hipStream_t stream) {
  (void)in_sizes; (void)n_in; (void)out_size; (void)ws_size;
  const float* hidden = (const float*)d_in[0];
  const float* mask   = (const float*)d_in[1];
  const float* conv_w = (const float*)d_in[2];
  const float* conv_b = (const float*)d_in[3];
  const float* out_w  = (const float*)d_in[4];
  const float* out_b  = (const float*)d_in[5];
  float* out = (float*)d_out;

  // d_out layout (floats): acoustic[16,2000,512] | token[16] | alphas[32000] | peak[32000]
  float* o_acoustic = out;
  float* o_token    = out + 16384000;
  float* o_alphas   = out + 16384016;
  float* o_peak     = out + 16416016;

  // workspace layout
  const size_t XPAD_ELEMS = (size_t)BB * TP * DD;   // 16,400,384 bf16
  const size_t WB_ELEMS   = (size_t)DD * KDIM;      //    786,432 bf16
  unsigned short* Xpad = (unsigned short*)d_ws;
  unsigned short* Wb   = Xpad + XPAD_ELEMS;
  float* alpha_acc     = (float*)(Wb + WB_ELEMS);   // 32,000 f32

  // K0: zero outputs that are sparsely written + the alpha accumulator
  k_zero<<<4096, 256, 0, stream>>>(o_acoustic, 16384016); // acoustic + token
  k_zero<<<64,   256, 0, stream>>>(alpha_acc, MROWS);

  // K1/K2: precision + layout transforms
  k_xpad <<<8192, 256, 0, stream>>>(hidden, Xpad);
  k_wprep<<<1536, 256, 0, stream>>>(conv_w, Wb);

  // K3: the WMMA GEMM (conv + relu + fused out_w dot), LDS-staged weights
  dim3 g(MROWS / 128, DD / 64);
  size_t lds_bytes = (size_t)2 * 64 * KCP * sizeof(unsigned short); // 34,816 B
  k_gemm<<<g, 256, lds_bytes, stream>>>(Xpad, Wb, conv_b, out_w, alpha_acc);

  // K4: alphas + token_num
  k_alpha<<<(MROWS + 255) / 256, 256, 0, stream>>>(alpha_acc, mask, out_b,
                                                   o_alphas, o_token);

  // K5: sequential CIF scan with on-the-fly compaction
  k_cif<<<BB, 512, 0, stream>>>(hidden, o_alphas, o_acoustic, o_peak);
}
